// TimeRecurrentCell_89369679495766
// MI455X (gfx1250) — compile-verified, matrix operand order checked by reading, not run on hardware
//
#include <hip/hip_runtime.h>

typedef __attribute__((ext_vector_type(16))) __bf16 v16bf;
typedef __attribute__((ext_vector_type(8)))  float  v8f;

#define T_STEPS 2048
#define B_SZ    64
#define D_SZ    512
#define H_SZ    512
#define K_SZ    1024          // D + H
#define NWG     32            // one workgroup per 16-wide hidden slice
#define NTHREADS 512          // 16 waves
#define W_PAIRS (2048u * 1024u / 2u)   // packed weight bf16 pairs

#define LOG2E 1.4426950408889634f

// fp32 -> bf16, round-to-nearest-even
__device__ __forceinline__ unsigned short f2bf(float f) {
  unsigned u = __builtin_bit_cast(unsigned, f);
  u += 0x7FFFu + ((u >> 16) & 1u);
  return (unsigned short)(u >> 16);
}

// K index of bf16 slot j (0..15) within a lane, per ISA 16-bit A/B VGPR layout:
// lanes 0-15: VGPR0-3 hold K=0..7, VGPR4-7 hold K=16..23; lanes 16-31: +8.
__device__ __forceinline__ int kidx(int lane, int j) {
  return ((j < 8) ? j : (j + 8)) + (((lane >> 4) & 1) << 3);
}

// Branch-free activations on hardware transcendentals (v_exp_f32 / v_rcp_f32).
// Saturation at +/-inf is exact: rcp(inf) == 0.
__device__ __forceinline__ float fast_sigmoid(float x) {
  return __builtin_amdgcn_rcpf(1.0f + __builtin_amdgcn_exp2f(-LOG2E * x));
}
__device__ __forceinline__ float fast_tanh(float x) {
  return 1.0f - 2.0f * __builtin_amdgcn_rcpf(
                          1.0f + __builtin_amdgcn_exp2f((2.0f * LOG2E) * x));
}

// ---------------------------------------------------------------- init
__global__ void lstm_init(float* __restrict__ cslot, unsigned* __restrict__ bar) {
  int i = blockIdx.x * blockDim.x + threadIdx.x;
  if (i < B_SZ * H_SZ) cslot[i] = 0.0f;   // c0 = 0
  if (i == 0) { bar[0] = 0u; bar[1] = 0u; }
}

// ------------------------------------------------ pack weights fp32 -> bf16
// Concatenated weight Wcat[n][k] = (k < D) ? W_ih[n][k] : W_hh[n][k-D].
// Packed into WMMA-B per-lane layout: blocks a = ntile*32 + kb, each block is
// 32 lanes x 16 bf16; GEMM loads v16bf at index a*32 + lane.
__global__ void lstm_pack_w(const float* __restrict__ Wih,
                            const float* __restrict__ Whh,
                            unsigned* __restrict__ wpk_u32) {
  int p = blockIdx.x * blockDim.x + threadIdx.x;  // one bf16 pair per thread
  int s = p << 1;
  int a = s >> 9;            // block: ntile*32 + kb
  int l = (s >> 4) & 31;     // lane
  int j = s & 15;            // even slot
  int ntile = a >> 5, kb = a & 31;
  int n = ntile * 16 + (l & 15);
  int k = kb * 32 + kidx(l, j);     // k even -> pair stays in one source matrix
  float w0, w1;
  if (k < D_SZ) {
    const float* pw = Wih + (size_t)n * D_SZ + k;
    w0 = pw[0]; w1 = pw[1];
  } else {
    const float* pw = Whh + (size_t)n * H_SZ + (k - D_SZ);
    w0 = pw[0]; w1 = pw[1];
  }
  wpk_u32[p] = (unsigned)f2bf(w0) | ((unsigned)f2bf(w1) << 16);
}

// ---------------------------------------------------------- persistent LSTM
__launch_bounds__(NTHREADS)
__global__ void lstm_persistent(const float* __restrict__ x,
                                const float* __restrict__ bih,
                                const float* __restrict__ bhh,
                                const v16bf* __restrict__ wv,
                                float* __restrict__ out,
                                unsigned* __restrict__ bar) {
  extern __shared__ v16bf dynsmem[];                  // A: 128 blocks * 32 lanes
  unsigned* smemAu = (unsigned*)dynsmem;              // 131072 B as u32 pairs
  float*    smemG  = (float*)((char*)dynsmem + 131072); // 16 tiles * 256 f32

  const int tid  = threadIdx.x;
  const int lane = tid & 31;
  const int w    = tid >> 5;     // wave 0..15
  const int mt   = w & 3;        // batch tile 0..3
  const int q    = w >> 2;       // gate 0..3 (i,f,g,o)
  const int jt   = blockIdx.x;   // hidden slice 0..31
  const int ntile = q * 32 + jt; // n-tile in [0,128): n = q*512 + jt*16 + ...

  float* cst = out + (size_t)T_STEPS * B_SZ * H_SZ + (size_t)B_SZ * H_SZ; // c state
  float* hT  = out + (size_t)T_STEPS * B_SZ * H_SZ;                        // h_T slot

  // Per-thread elementwise elements: e0 = tid, e1 = tid + 512 over (b, jl) pairs.
  const int e0 = tid,        e1 = tid + NTHREADS;
  const int b0 = e0 >> 4,    j0 = e0 & 15,  h0 = jt * 16 + j0;
  const int b1 = e1 >> 4,    j1 = e1 & 15,  h1 = jt * 16 + j1;
  const float bi0 = bih[0*H_SZ+h0] + bhh[0*H_SZ+h0];
  const float bf0 = bih[1*H_SZ+h0] + bhh[1*H_SZ+h0];
  const float bg0 = bih[2*H_SZ+h0] + bhh[2*H_SZ+h0];
  const float bo0 = bih[3*H_SZ+h0] + bhh[3*H_SZ+h0];
  const float bi1 = bih[0*H_SZ+h1] + bhh[0*H_SZ+h1];
  const float bf1 = bih[1*H_SZ+h1] + bhh[1*H_SZ+h1];
  const float bg1 = bih[2*H_SZ+h1] + bhh[2*H_SZ+h1];
  const float bo1 = bih[3*H_SZ+h1] + bhh[3*H_SZ+h1];

  for (int t = 0; t < T_STEPS; ++t) {
    // ---- pack A = [x_t | h_{t-1}] into LDS in WMMA A layout (bf16) ----
    const float* xrow  = x + (size_t)t * B_SZ * D_SZ;
    const float* hprev = out + (size_t)(t - 1) * B_SZ * H_SZ; // guarded for t==0
    for (int p = tid; p < 32768; p += NTHREADS) {
      int s = p << 1;
      int a = s >> 9;            // block: mtile*32 + kb
      int l = (s >> 4) & 31;
      int j = s & 15;
      int amt = a >> 5, akb = a & 31;
      int m = amt * 16 + (l & 15);
      int k = akb * 32 + kidx(l, j);
      float v0, v1;
      if (k < D_SZ) {
        const float* px = xrow + (size_t)m * D_SZ + k;
        v0 = px[0]; v1 = px[1];
      } else if (t == 0) {
        v0 = 0.0f; v1 = 0.0f;
      } else {
        const float* ph = hprev + (size_t)m * H_SZ + (k - D_SZ);
        v0 = ph[0]; v1 = ph[1];
      }
      smemAu[p] = (unsigned)f2bf(v0) | ((unsigned)f2bf(v1) << 16);
    }
    __syncthreads();

    // ---- GEMM: 16x16 gate tile, K=1024 via 32 x wmma_f32_16x16x32_bf16 ----
    v8f acc = {0.f, 0.f, 0.f, 0.f, 0.f, 0.f, 0.f, 0.f};
#pragma unroll 4
    for (int kb = 0; kb < 32; ++kb) {
      v16bf av = dynsmem[(mt * 32 + kb) * 32 + lane];               // LDS
      v16bf bv = wv[((size_t)ntile * 32 + kb) * 32 + lane];         // L2-resident
      acc = __builtin_amdgcn_wmma_f32_16x16x32_bf16(
          false, av, false, bv, (short)0, acc, false, false);
    }

    // ---- stash gate tile to LDS (C/D layout: lanes16-31 hold M=8..15) ----
    {
      int mloc = (lane >> 4) << 3;
      int nloc = lane & 15;
      float* g = smemG + (q * 4 + mt) * 256 + nloc;
#pragma unroll
      for (int r = 0; r < 8; ++r) g[(mloc + r) * 16] = acc[r];
    }
    __syncthreads();

    // ---- LSTM elementwise (torch gate order i,f,g,o), 2 elems/thread ----
    {
      int mt0 = b0 >> 4, ml0 = b0 & 15;
      float gi = fast_sigmoid(smemG[(0*4+mt0)*256 + ml0*16 + j0] + bi0);
      float gf = fast_sigmoid(smemG[(1*4+mt0)*256 + ml0*16 + j0] + bf0);
      float gg = fast_tanh   (smemG[(2*4+mt0)*256 + ml0*16 + j0] + bg0);
      float go = fast_sigmoid(smemG[(3*4+mt0)*256 + ml0*16 + j0] + bo0);
      float cp = cst[(size_t)b0 * H_SZ + h0];
      float cn = gf * cp + gi * gg;
      float hn = go * fast_tanh(cn);
      cst[(size_t)b0 * H_SZ + h0] = cn;
      out[(size_t)t * B_SZ * H_SZ + (size_t)b0 * H_SZ + h0] = hn;
      if (t == T_STEPS - 1) hT[(size_t)b0 * H_SZ + h0] = hn;
    }
    {
      int mt1 = b1 >> 4, ml1 = b1 & 15;
      float gi = fast_sigmoid(smemG[(0*4+mt1)*256 + ml1*16 + j1] + bi1);
      float gf = fast_sigmoid(smemG[(1*4+mt1)*256 + ml1*16 + j1] + bf1);
      float gg = fast_tanh   (smemG[(2*4+mt1)*256 + ml1*16 + j1] + bg1);
      float go = fast_sigmoid(smemG[(3*4+mt1)*256 + ml1*16 + j1] + bo1);
      float cp = cst[(size_t)b1 * H_SZ + h1];
      float cn = gf * cp + gi * gg;
      float hn = go * fast_tanh(cn);
      cst[(size_t)b1 * H_SZ + h1] = cn;
      out[(size_t)t * B_SZ * H_SZ + (size_t)b1 * H_SZ + h1] = hn;
      if (t == T_STEPS - 1) hT[(size_t)b1 * H_SZ + h1] = hn;
    }

    // ---- device-wide barrier: publish h/c, then next step may read ----
    __threadfence();
    __syncthreads();
    if (tid == 0) {
      unsigned gsn = __hip_atomic_load(&bar[1], __ATOMIC_RELAXED,
                                       __HIP_MEMORY_SCOPE_AGENT);
      unsigned arr = __hip_atomic_fetch_add(&bar[0], 1u, __ATOMIC_ACQ_REL,
                                            __HIP_MEMORY_SCOPE_AGENT);
      if (arr == NWG - 1) {
        __hip_atomic_store(&bar[0], 0u, __ATOMIC_RELAXED,
                           __HIP_MEMORY_SCOPE_AGENT);
        __hip_atomic_fetch_add(&bar[1], 1u, __ATOMIC_ACQ_REL,
                               __HIP_MEMORY_SCOPE_AGENT);
      } else {
        while (__hip_atomic_load(&bar[1], __ATOMIC_ACQUIRE,
                                 __HIP_MEMORY_SCOPE_AGENT) == gsn) {
          __builtin_amdgcn_s_sleep(1);
        }
      }
      __threadfence();
    }
    __syncthreads();
  }
}

// ---------------------------------------------------------------- launcher
extern "C" void kernel_launch(void* const* d_in, const int* in_sizes, int n_in,
                              void* d_out, int out_size, void* d_ws, size_t ws_size,
                              hipStream_t stream) {
  (void)in_sizes; (void)n_in; (void)out_size; (void)ws_size;
  const float* x   = (const float*)d_in[0];
  const float* Wih = (const float*)d_in[1];
  const float* Whh = (const float*)d_in[2];
  const float* bih = (const float*)d_in[3];
  const float* bhh = (const float*)d_in[4];
  float* out = (float*)d_out;

  unsigned* wpk_u32 = (unsigned*)d_ws;                       // 4 MB packed bf16 weights
  unsigned* bar     = (unsigned*)((char*)d_ws + (4u << 20)); // barrier counter + gen
  float* cslot = out + (size_t)T_STEPS * B_SZ * H_SZ + (size_t)B_SZ * H_SZ;

  lstm_init<<<(B_SZ * H_SZ + 255) / 256, 256, 0, stream>>>(cslot, bar);
  lstm_pack_w<<<W_PAIRS / 256, 256, 0, stream>>>(Wih, Whh, wpk_u32);
  lstm_persistent<<<NWG, NTHREADS, 147456, stream>>>(
      x, bih, bhh, (const v16bf*)wpk_u32, out, bar);
}
